// MultiHeadAttention_6966436954703
// MI455X (gfx1250) — matrix operand order repaired and checked by reading, not compiled, not run on hardware
//
#include <hip/hip_runtime.h>

#define Bs 2
#define Ts 4096
#define Ds 512
#define Hs 8
#define HDs 64

typedef __bf16 bf16_t;
typedef __attribute__((ext_vector_type(8)))  __bf16 bf16x8;
typedef __attribute__((ext_vector_type(16))) __bf16 v16bf;
typedef __attribute__((ext_vector_type(8)))  float  v8f;
typedef __attribute__((ext_vector_type(4)))  int    v4i;

union V16u { v16bf v; bf16x8 h[2]; };
union PKu  { bf16x8 h; int i[4]; };

#if __has_builtin(__builtin_amdgcn_global_load_async_to_lds_b128) && \
    __has_builtin(__builtin_amdgcn_s_wait_asynccnt)
#define HAVE_ASYNC 1
#else
#define HAVE_ASYNC 0
#endif

__device__ __forceinline__ v8f wmma_bf16(v16bf a, v16bf b, v8f c) {
  // D(f32 16x16) = A(bf16 16x32) * B(bf16 32x16) + C
  return __builtin_amdgcn_wmma_f32_16x16x32_bf16(false, a, false, b, (short)0, c, false, false);
}

// B-fragment (K x 16, 16-bit): lane n holds 16 contiguous K values
__device__ __forceinline__ v16bf ld_bfrag(const bf16_t* p) {
  V16u u;
  u.h[0] = *(const bf16x8*)(p);
  u.h[1] = *(const bf16x8*)(p + 8);
  return u.v;
}
// A-fragment (16 x 32, 16-bit): lane holds K = 2*kb..2*kb+15 contiguous (kb = 0 or 8)
__device__ __forceinline__ v16bf ld_afrag(const bf16_t* p, int kb) {
  V16u u;
  u.h[0] = *(const bf16x8*)(p + 2 * kb);
  u.h[1] = *(const bf16x8*)(p + 2 * kb + 8);
  return u.v;
}

// 16-byte global -> LDS copy, async (ASYNCcnt) when available
__device__ __forceinline__ void async_copy16(const bf16_t* g, bf16_t* l) {
#if HAVE_ASYNC
  __builtin_amdgcn_global_load_async_to_lds_b128(
      (__attribute__((address_space(1))) v4i*)(void*)g,
      (__attribute__((address_space(3))) v4i*)l, 0, 0);
#else
  *(bf16x8*)l = *(const bf16x8*)g;
#endif
}
__device__ __forceinline__ void wait_async() {
#if HAVE_ASYNC
  __builtin_amdgcn_s_wait_asynccnt(0);
#endif
}

__global__ void cvt_f32_bf16(const float* __restrict__ in, bf16_t* __restrict__ out, int n) {
  int i = blockIdx.x * blockDim.x + threadIdx.x;
  if (i < n) out[i] = (bf16_t)in[i];
}

// C[m,n] = sum_k X[m,k] * W[n,k]   (X: [8192,512] bf16 rowmajor, W: [512,512] bf16 rowmajor)
// MODE 0: store bf16 [B,H,T,HD]; MODE 1: store bf16 [B,H,HD,T]; MODE 2: store f32 [M,N] + bias
template<int MODE>
__global__ __launch_bounds__(256) void gemm_bf16_wmma(
    const bf16_t* __restrict__ X, const bf16_t* __restrict__ W,
    bf16_t* __restrict__ outb, float* __restrict__ outf, const float* __restrict__ bias)
{
  const int lane = threadIdx.x & 31;
  const int wave = threadIdx.x >> 5;
  const int wm = blockIdx.x * 128 + (wave >> 1) * 32;
  const int wn = blockIdx.y * 64  + (wave & 1) * 32;
  const int l15 = lane & 15;
  const int kbA = (lane & 16) ? 8 : 0;
  const int kbB = (lane & 16) ? 16 : 0;

  v8f acc[2][2] = {};

  for (int kk = 0; kk < Ds; kk += 32) {
    v16bf a0 = ld_afrag(X + (size_t)(wm + l15) * Ds + kk, kbA);
    v16bf a1 = ld_afrag(X + (size_t)(wm + 16 + l15) * Ds + kk, kbA);
    v16bf b0 = ld_bfrag(W + (size_t)(wn + l15) * Ds + kk + kbB);
    v16bf b1 = ld_bfrag(W + (size_t)(wn + 16 + l15) * Ds + kk + kbB);
    acc[0][0] = wmma_bf16(a0, b0, acc[0][0]);
    acc[0][1] = wmma_bf16(a0, b1, acc[0][1]);
    acc[1][0] = wmma_bf16(a1, b0, acc[1][0]);
    acc[1][1] = wmma_bf16(a1, b1, acc[1][1]);
  }

  const int mh = (lane & 16) ? 8 : 0;
  #pragma unroll
  for (int mc = 0; mc < 2; ++mc)
  #pragma unroll
  for (int nc = 0; nc < 2; ++nc)
  #pragma unroll
  for (int r = 0; r < 8; ++r) {
    int m = wm + mc * 16 + mh + r;      // global row = b*T + t
    int n = wn + nc * 16 + l15;         // global col = h*HD + hd
    float v = acc[mc][nc][r];
    if (MODE == 0) {
      int b = m >> 12, t = m & (Ts - 1), h = n >> 6, hd = n & 63;
      outb[(((size_t)b * Hs + h) * Ts + t) * HDs + hd] = (bf16_t)v;
    } else if (MODE == 1) {
      int b = m >> 12, t = m & (Ts - 1), h = n >> 6, hd = n & 63;
      outb[(((size_t)b * Hs + h) * HDs + hd) * Ts + t] = (bf16_t)v;
    } else {
      outf[(size_t)m * Ds + n] = v + bias[n];
    }
  }
}

// Stage one 64-key tile: K (64 rows x 64 d, 8KB) and V^T (64 d x 64 keys, 8KB) into LDS.
// 256 threads x 2 x 16B for each of K and V.
__device__ __forceinline__ void stage_tile(const bf16_t* __restrict__ Kb,
                                           const bf16_t* __restrict__ Vb,
                                           bf16_t* sk, bf16_t* sv,
                                           int tid, int kbase) {
  #pragma unroll
  for (int s = 0; s < 2; ++s) {
    int c = tid + s * 256;             // 0..511 -> 512 x 16B chunks
    int row = c >> 3;                  // 0..63
    int col = (c & 7) * 8;             // 0..56
    async_copy16(Kb + (size_t)(kbase + row) * HDs + col, sk + row * 64 + col);
    async_copy16(Vb + (size_t)row * Ts + kbase + col,    sv + row * 64 + col);
  }
}

// One 64-key tile of the online-softmax attention for one wave (16 queries).
// S^T = K*Q^T so each lane owns one query column; O^T = V^T*P^T accumulated in regs.
template<bool MASK>
__device__ __forceinline__ void attn_tile(const bf16_t* sk, const bf16_t* sv,
                                          v16bf q0, v16bf q1, int kbase, int iq,
                                          float& mrun, float& lrun, v8f (&o)[4])
{
  const int lane  = threadIdx.x & 31;
  const int l15   = lane & 15;
  const int half8 = (lane & 16) ? 8 : 0;
  const int kbA   = (lane & 16) ? 8 : 0;
  const float SC  = 0.125f * 1.44269504088896f;   // 1/sqrt(HD) * log2(e)

  float p[4][8];
  #pragma unroll
  for (int jc = 0; jc < 4; ++jc) {
    const bf16_t* kp = sk + (jc * 16 + l15) * 64;
    v16bf a0 = ld_afrag(kp, kbA);
    v16bf a1 = ld_afrag(kp + 32, kbA);
    v8f c = {};
    c = wmma_bf16(a0, q0, c);
    c = wmma_bf16(a1, q1, c);
    #pragma unroll
    for (int r = 0; r < 8; ++r) {
      float v = c[r] * SC;
      if (MASK) {
        int j = kbase + jc * 16 + half8 + r;
        v = (j > iq) ? -1.0e30f : v;
      }
      p[jc][r] = v;
    }
  }

  float tm = -1.0e30f;
  #pragma unroll
  for (int jc = 0; jc < 4; ++jc)
    #pragma unroll
    for (int r = 0; r < 8; ++r) tm = fmaxf(tm, p[jc][r]);
  tm = fmaxf(tm, __shfl_xor(tm, 16, 32));
  const float mnew  = fmaxf(mrun, tm);
  const float alpha = exp2f(mrun - mnew);
  float ls = 0.0f;
  #pragma unroll
  for (int jc = 0; jc < 4; ++jc)
    #pragma unroll
    for (int r = 0; r < 8; ++r) { float e = exp2f(p[jc][r] - mnew); p[jc][r] = e; ls += e; }
  ls += __shfl_xor(ls, 16, 32);
  lrun = lrun * alpha + ls;
  mrun = mnew;
  #pragma unroll
  for (int f = 0; f < 4; ++f)
    #pragma unroll
    for (int r = 0; r < 8; ++r) o[f][r] *= alpha;

  bf16x8 pk[4];
  #pragma unroll
  for (int jc = 0; jc < 4; ++jc)
    #pragma unroll
    for (int r = 0; r < 8; ++r) pk[jc][r] = (bf16_t)p[jc][r];

  #pragma unroll
  for (int jk = 0; jk < 2; ++jk) {
    PKu snd; snd.h = (lane < 16) ? pk[2 * jk + 1] : pk[2 * jk];
    PKu rcv;
    #pragma unroll
    for (int w = 0; w < 4; ++w) rcv.i[w] = __shfl_xor(snd.i[w], 16, 32);
    V16u bp;
    if (lane < 16) { bp.h[0] = pk[2 * jk]; bp.h[1] = rcv.h; }
    else           { bp.h[0] = rcv.h;      bp.h[1] = pk[2 * jk + 1]; }
    #pragma unroll
    for (int f = 0; f < 4; ++f) {
      const bf16_t* vp = sv + (f * 16 + l15) * 64 + jk * 32;
      v16bf av = ld_afrag(vp, kbA);
      o[f] = wmma_bf16(av, bp.v, o[f]);
    }
  }
}

__global__ __launch_bounds__(256) void flash_attn_wmma(
    const bf16_t* __restrict__ Q, const bf16_t* __restrict__ K,
    const bf16_t* __restrict__ Vt, bf16_t* __restrict__ ctx)
{
  __shared__ __align__(16) bf16_t sK[2][64 * 64];
  __shared__ __align__(16) bf16_t sV[2][64 * 64];

  const int tid   = threadIdx.x;
  const int lane  = tid & 31;
  const int wave  = tid >> 5;
  const int bh    = blockIdx.y;                  // b*H + h
  const int qbase = blockIdx.x * 128;
  const int qrow  = qbase + wave * 16;
  const int l15   = lane & 15;
  const int half8 = (lane & 16) ? 8 : 0;
  const int iq    = qrow + l15;

  const bf16_t* Qb = Q  + (size_t)bh * Ts * HDs;
  const bf16_t* Kb = K  + (size_t)bh * Ts * HDs;
  const bf16_t* Vb = Vt + (size_t)bh * HDs * Ts;

  // Q as B-fragments for the two 32-wide d chunks; reused for every key tile.
  const int kbB = (lane & 16) ? 16 : 0;
  v16bf q0 = ld_bfrag(Qb + (size_t)iq * HDs + kbB);
  v16bf q1 = ld_bfrag(Qb + (size_t)iq * HDs + 32 + kbB);

  float mrun = -1.0e30f;
  float lrun = 0.0f;
  v8f o[4] = {};

  const int nt = (qbase >> 6) + 2;   // block-uniform tile count (covers wave 7's diagonal)
  const int mt = qrow >> 6;          // the single tile straddling this wave's diagonal

  stage_tile(Kb, Vb, sK[0], sV[0], tid, 0);
  wait_async();
  __syncthreads();

  for (int t = 0; t < nt; ++t) {
    const int buf = t & 1;
    if (t + 1 < nt) stage_tile(Kb, Vb, sK[buf ^ 1], sV[buf ^ 1], tid, (t + 1) * 64);
    if (t < mt)       attn_tile<false>(sK[buf], sV[buf], q0, q1, t * 64, iq, mrun, lrun, o);
    else if (t == mt) attn_tile<true >(sK[buf], sV[buf], q0, q1, t * 64, iq, mrun, lrun, o);
    // t > mt: fully masked for this wave -> skip compute, still hit the barriers
    wait_async();
    __syncthreads();
  }

  // normalize and store ctx as bf16 [B, T, H*HD] (d contiguous per lane -> 16B stores)
  const float inv = 1.0f / lrun;
  const int bI = bh >> 3, hI = bh & 7;
  #pragma unroll
  for (int f = 0; f < 4; ++f) {
    bf16x8 r8;
    #pragma unroll
    for (int r = 0; r < 8; ++r) r8[r] = (bf16_t)(o[f][r] * inv);
    bf16_t* cp = ctx + ((size_t)bI * Ts + iq) * Ds + hI * HDs + f * 16 + half8;
    *(bf16x8*)cp = r8;
  }
}

extern "C" void kernel_launch(void* const* d_in, const int* in_sizes, int n_in,
                              void* d_out, int out_size, void* d_ws, size_t ws_size,
                              hipStream_t stream) {
  (void)in_sizes; (void)n_in; (void)out_size; (void)ws_size;
  const float* x  = (const float*)d_in[0];
  const float* Wq = (const float*)d_in[1];
  const float* Wk = (const float*)d_in[2];
  const float* Wv = (const float*)d_in[3];
  const float* Wo = (const float*)d_in[4];
  const float* bo = (const float*)d_in[5];
  float* out = (float*)d_out;

  const size_t NX = (size_t)Bs * Ts * Ds;   // 4,194,304 elems
  const size_t NW = (size_t)Ds * Ds;        // 262,144 elems
  const size_t NQ = (size_t)Bs * Hs * Ts * HDs;

  char* w = (char*)d_ws;
  bf16_t* xb  = (bf16_t*)w; w += NX * 2;
  bf16_t* wqb = (bf16_t*)w; w += NW * 2;
  bf16_t* wkb = (bf16_t*)w; w += NW * 2;
  bf16_t* wvb = (bf16_t*)w; w += NW * 2;
  bf16_t* wob = (bf16_t*)w; w += NW * 2;
  bf16_t* qb  = (bf16_t*)w; w += NQ * 2;
  bf16_t* kb  = (bf16_t*)w; w += NQ * 2;
  bf16_t* vtb = (bf16_t*)w; w += NQ * 2;
  bf16_t* cxb = (bf16_t*)w; w += NX * 2;

  cvt_f32_bf16<<<(int)((NX + 255) / 256), 256, 0, stream>>>(x,  xb,  (int)NX);
  cvt_f32_bf16<<<(int)((NW + 255) / 256), 256, 0, stream>>>(Wq, wqb, (int)NW);
  cvt_f32_bf16<<<(int)((NW + 255) / 256), 256, 0, stream>>>(Wk, wkb, (int)NW);
  cvt_f32_bf16<<<(int)((NW + 255) / 256), 256, 0, stream>>>(Wv, wvb, (int)NW);
  cvt_f32_bf16<<<(int)((NW + 255) / 256), 256, 0, stream>>>(Wo, wob, (int)NW);

  dim3 gg(Bs * Ts / 128, Ds / 64);
  gemm_bf16_wmma<0><<<gg, 256, 0, stream>>>(xb, wqb, qb,  nullptr, nullptr);
  gemm_bf16_wmma<0><<<gg, 256, 0, stream>>>(xb, wkb, kb,  nullptr, nullptr);
  gemm_bf16_wmma<1><<<gg, 256, 0, stream>>>(xb, wvb, vtb, nullptr, nullptr);

  dim3 ga(Ts / 128, Bs * Hs);
  flash_attn_wmma<<<ga, 256, 0, stream>>>(qb, kb, vtb, cxb);

  gemm_bf16_wmma<2><<<gg, 256, 0, stream>>>(cxb, wob, nullptr, out, bo);
}
